// StyleDecorator_48541720379574
// MI455X (gfx1250) — matrix operand order repaired and checked by reading, not compiled, not run on hardware
//
#include <hip/hip_runtime.h>
#include <hip/hip_bf16.h>
#include <utility>

// ---------------- problem constants (B,C,H,W fixed by setup_inputs) ---------
static constexpr int B  = 2;
static constexpr int C  = 256;
static constexpr int H  = 48;
static constexpr int W  = 48;
static constexpr int N  = H * W;        // 2304 spatial positions
static constexpr int KK = C * 9;        // 2304 patch-vector length (p=3)
static constexpr int NS_ITERS = 14;     // Newton-Schulz iterations

typedef __attribute__((ext_vector_type(2)))  float     v2f;
typedef __attribute__((ext_vector_type(8)))  float     v8f;
typedef __attribute__((ext_vector_type(8)))  _Float16  v8h;
typedef __attribute__((ext_vector_type(16))) _Float16  v16h;

// ---------------- WMMA wrappers --------------------------------------------
__device__ __forceinline__ v8f wmma_f32_k4(v2f a, v2f b, v8f c) {
    // V_WMMA_F32_16X16X4_F32 : A 16x4 f32, B 4x16 f32, C/D 16x16 f32
    return __builtin_amdgcn_wmma_f32_16x16x4_f32(false, a, false, b,
                                                 (short)0, c, false, false);
}
__device__ __forceinline__ v8f wmma_f16_k32(v16h a, v16h b, v8f c) {
    // V_WMMA_F32_16X16X32_F16
    return __builtin_amdgcn_wmma_f32_16x16x32_f16(false, a, false, b,
                                                  (short)0, c, false, false);
}

// C/D 16x16 f32 layout: lane L -> column = L%16 ; VGPR r -> row = r + 8*(L/16)
__device__ __forceinline__ void store_tile(float* D, int ldd, int row0, int col0,
                                           v8f acc, float scale) {
    int lane = threadIdx.x & 31;
    int col  = col0 + (lane & 15);
    int mb   = row0 + ((lane >> 4) << 3);
#pragma unroll
    for (int r = 0; r < 8; ++r) D[(size_t)(mb + r) * ldd + col] = acc[r] * scale;
}

// A 16x32 f16 fragment load: lanes 0-15 hold K 0-7 & 16-23, lanes 16-31 hold
// K 8-15 & 24-31 of row M = lane%16.  ap must already include the per-lane
// row and the (lane>>4)*8 K offset.
__device__ __forceinline__ v16h load_afrag(const _Float16* ap) {
    v8h lo = *(const v8h*)(ap);
    v8h hi = *(const v8h*)(ap + 16);
    return __builtin_shufflevector(lo, hi, 0, 1, 2, 3, 4, 5, 6, 7,
                                   8, 9, 10, 11, 12, 13, 14, 15);
}

// ---------------- per-(b,c) mean-center (also used in-place) ----------------
__global__ void k_center(const float* __restrict__ src, float* __restrict__ dst,
                         float* __restrict__ meanOut) {
    int bc = blockIdx.x;                       // b*C + c
    const float* s = src + (size_t)bc * N;
    float*       d = dst + (size_t)bc * N;
    __shared__ float red[256];
    float sum = 0.f;
    for (int j = threadIdx.x; j < N; j += 256) sum += s[j];
    red[threadIdx.x] = sum; __syncthreads();
    for (int st = 128; st > 0; st >>= 1) {
        if (threadIdx.x < st) red[threadIdx.x] += red[threadIdx.x + st];
        __syncthreads();
    }
    float mean = red[0] * (1.f / N);
    for (int j = threadIdx.x; j < N; j += 256) d[j] = s[j] - mean;
    if (threadIdx.x == 0 && meanOut) meanOut[bc] = mean;
}

// ---------------- covariance: Cov = F F^T / (N-1)  (f32 WMMA) ---------------
__global__ void k_cov(const float* __restrict__ F, float* __restrict__ Cov) {
    int b = blockIdx.z;
    const float* Fb = F   + (size_t)b * C * N;
    float*       Cb = Cov + (size_t)b * C * C;
    int row0 = blockIdx.y * 16, col0 = blockIdx.x * 16;
    int lane = threadIdx.x & 31;
    int m = row0 + (lane & 15);
    int n = col0 + (lane & 15);
    int ko = (lane >> 4) << 1;                 // 0 or 2
    v8f acc = {};
    for (int k = 0; k < N; k += 4) {
        v2f a = *(const v2f*)(Fb + (size_t)m * N + k + ko);   // A 16x4 frag
        v2f bb = *(const v2f*)(Fb + (size_t)n * N + k + ko);  // B = F^T column
        acc = wmma_f32_k4(a, bb, acc);
    }
    store_tile(Cb, C, row0, col0, acc, 1.f / (float)(N - 1));
}

// ---------------- trace + derived scalars per batch -------------------------
// sc[b*4+{0,1,2,3}] = {tr, 1/tr, 1/sqrt(tr), sqrt(tr)}
__global__ void k_trace(const float* __restrict__ Cov, float* __restrict__ sc) {
    int b = blockIdx.x;
    __shared__ float red[256];
    red[threadIdx.x] = Cov[(size_t)b * C * C + (size_t)threadIdx.x * (C + 1)];
    __syncthreads();
    for (int st = 128; st > 0; st >>= 1) {
        if (threadIdx.x < st) red[threadIdx.x] += red[threadIdx.x + st];
        __syncthreads();
    }
    if (threadIdx.x == 0) {
        float c = red[0];
        sc[b * 4 + 0] = c;
        sc[b * 4 + 1] = 1.f / c;
        sc[b * 4 + 2] = rsqrtf(c);
        sc[b * 4 + 3] = sqrtf(c);
    }
}

// ---------------- Newton-Schulz init: Y = A/tr, Z = I -----------------------
__global__ void k_nsinit(const float* __restrict__ Cov, const float* __restrict__ sc,
                         float* __restrict__ Y, float* __restrict__ Z) {
    size_t i = (size_t)blockIdx.x * 256 + threadIdx.x;   // B*C*C total
    int b = (int)(i >> 16);
    int e = (int)(i & 65535);
    Y[i] = Cov[i] * sc[b * 4 + 1];
    Z[i] = ((e >> 8) == (e & 255)) ? 1.f : 0.f;
}

// ---------------- T = 1.5 I - 0.5 * (Z @ Y)  (256x256, f32 WMMA) ------------
__global__ void k_nsT(const float* __restrict__ Z, const float* __restrict__ Y,
                      float* __restrict__ T) {
    int b = blockIdx.z;
    const float* Zb = Z + (size_t)b * C * C;
    const float* Yb = Y + (size_t)b * C * C;
    float*       Tb = T + (size_t)b * C * C;
    int row0 = blockIdx.y * 16, col0 = blockIdx.x * 16;
    int lane = threadIdx.x & 31;
    int m = row0 + (lane & 15), n = col0 + (lane & 15);
    int ko = (lane >> 4) << 1;
    v8f acc = {};
    for (int k = 0; k < C; k += 4) {
        v2f a = *(const v2f*)(Zb + (size_t)m * C + k + ko);
        v2f bb; bb.x = Yb[(size_t)(k + ko) * C + n];
                bb.y = Yb[(size_t)(k + ko + 1) * C + n];
        acc = wmma_f32_k4(a, bb, acc);
    }
    int col = col0 + (lane & 15);
    int mb  = row0 + ((lane >> 4) << 3);
#pragma unroll
    for (int r = 0; r < 8; ++r) {
        int row = mb + r;
        Tb[(size_t)row * C + col] = ((row == col) ? 1.5f : 0.f) - 0.5f * acc[r];
    }
}

// ---------------- D = A @ B   (256x256x256, f32 WMMA) -----------------------
__global__ void k_mm256(const float* __restrict__ A, const float* __restrict__ Bm,
                        float* __restrict__ D) {
    int b = blockIdx.z;
    const float* Ab = A  + (size_t)b * C * C;
    const float* Bb = Bm + (size_t)b * C * C;
    float*       Db = D  + (size_t)b * C * C;
    int row0 = blockIdx.y * 16, col0 = blockIdx.x * 16;
    int lane = threadIdx.x & 31;
    int m = row0 + (lane & 15), n = col0 + (lane & 15);
    int ko = (lane >> 4) << 1;
    v8f acc = {};
    for (int k = 0; k < C; k += 4) {
        v2f a = *(const v2f*)(Ab + (size_t)m * C + k + ko);
        v2f bb; bb.x = Bb[(size_t)(k + ko) * C + n];
                bb.y = Bb[(size_t)(k + ko + 1) * C + n];
        acc = wmma_f32_k4(a, bb, acc);
    }
    store_tile(Db, C, row0, col0, acc, 1.f);
}

// ---------------- D = scale * (M @ F) [+ tmean, blend] (256 x 2304) ---------
// MODE 0: whiten apply (scale = 1/sqrt(tr)).  MODE 1: color apply + blend.
template <int MODE>
__global__ void k_apply(const float* __restrict__ Mtx, const float* __restrict__ F,
                        const float* __restrict__ sc, int scIdx,
                        const float* __restrict__ tmean,
                        const float* __restrict__ content,
                        const float* __restrict__ strengthPtr,
                        float* __restrict__ D) {
    int b = blockIdx.z;
    const float* Mb = Mtx + (size_t)b * C * C;
    const float* Fb = F   + (size_t)b * C * N;
    int row0 = blockIdx.y * 16;                // channel
    int col0 = blockIdx.x * 16;                // spatial position
    int lane = threadIdx.x & 31;
    int m = row0 + (lane & 15), n = col0 + (lane & 15);
    int ko = (lane >> 4) << 1;
    v8f acc = {};
    for (int k = 0; k < C; k += 4) {
        v2f a = *(const v2f*)(Mb + (size_t)m * C + k + ko);
        v2f bb; bb.x = Fb[(size_t)(k + ko) * N + n];
                bb.y = Fb[(size_t)(k + ko + 1) * N + n];
        acc = wmma_f32_k4(a, bb, acc);
    }
    float scale = sc[b * 4 + scIdx];
    int col = col0 + (lane & 15);
    int mb  = row0 + ((lane >> 4) << 3);
    float st = (MODE == 1) ? *strengthPtr : 0.f;
#pragma unroll
    for (int r = 0; r < 8; ++r) {
        int row = mb + r;
        float v = acc[r] * scale;
        size_t oi = ((size_t)b * C + row) * N + col;
        if (MODE == 1) {
            v += tmean[b * C + row];                       // + style mean
            v = (1.f - st) * content[oi] + st * v;         // strength blend
        }
        D[oi] = v;
    }
}

// ---------------- per-position channel sum of squares -----------------------
__global__ void k_ssq(const float* __restrict__ ns, float* __restrict__ ssq) {
    int i = blockIdx.x * 256 + threadIdx.x;
    if (i >= B * N) return;
    int b = i / N, pos = i % N;
    const float* p = ns + (size_t)b * C * N + pos;
    float s = 0.f;
    for (int c = 0; c < C; ++c) { float v = p[(size_t)c * N]; s += v * v; }
    ssq[i] = s;
}

// ---------------- 1/knorm over 3x3 windows ----------------------------------
__global__ void k_knorm(const float* __restrict__ ssq, float* __restrict__ kinv) {
    int i = blockIdx.x * 256 + threadIdx.x;
    if (i >= B * N) return;
    int b = i / N, pos = i % N, y = pos / W, x = pos % W;
    float s = 0.f;
    for (int dy = -1; dy <= 1; ++dy)
        for (int dx = -1; dx <= 1; ++dx) {
            int yy = y + dy, xx = x + dx;
            if (yy >= 0 && yy < H && xx >= 0 && xx < W)
                s += ssq[b * N + yy * W + xx];
        }
    kinv[i] = rsqrtf(fmaxf(s, 1e-20f));
}

// ---------------- build f16 patch matrix  P[b][n][c*9 + d] ------------------
__global__ void k_patches(const float* __restrict__ f, _Float16* __restrict__ P) {
    int i = blockIdx.x * 256 + threadIdx.x;              // B*N*C threads
    if (i >= B * N * C) return;
    int b = i / (N * C);
    int rem = i % (N * C);
    int n = rem / C, c = rem % C;
    int y = n / W, x = n % W;
    const float* src = f + ((size_t)b * C + c) * N;
    _Float16* dst = P + ((size_t)b * N + n) * KK + (size_t)c * 9;
#pragma unroll
    for (int di = 0; di < 3; ++di)
#pragma unroll
        for (int dj = 0; dj < 3; ++dj) {
            int yy = y + di - 1, xx = x + dj - 1;
            float v = (yy >= 0 && yy < H && xx >= 0 && xx < W)
                          ? src[yy * W + xx] : 0.f;
            dst[di * 3 + dj] = (_Float16)v;
        }
}

// ---------------- score GEMM (f16 WMMA, 2x2 register-blocked) ---------------
// Each wave computes a 32(n) x 32(m) tile of score[n,m] with 4 accumulators,
// reusing each A/B fragment twice, then folds in 1/knorm and does a packed
// (score,idx) u64 atomicMax argmax per output column.
__global__ void k_score(const _Float16* __restrict__ PS,
                        const _Float16* __restrict__ PC,
                        const float* __restrict__ kinv,
                        unsigned long long* __restrict__ keys) {
    int b  = blockIdx.z;
    int tn = blockIdx.y * 32;                  // style rows   (n)
    int tm = blockIdx.x * 32;                  // content cols (m)
    int lane = threadIdx.x & 31;
    const _Float16* Sb = PS + (size_t)b * N * KK;
    const _Float16* Cb = PC + (size_t)b * N * KK;
    const _Float16* ap0 = Sb + (size_t)(tn + (lane & 15)) * KK + ((lane >> 4) << 3);
    const _Float16* ap1 = ap0 + (size_t)16 * KK;
    const _Float16* bp0 = Cb + (size_t)(tm + (lane & 15)) * KK + ((lane >> 4) << 3);
    const _Float16* bp1 = bp0 + (size_t)16 * KK;
    v8f a00 = {}, a01 = {}, a10 = {}, a11 = {};
    for (int k = 0; k < KK; k += 32) {
        v16h A0 = load_afrag(ap0 + k);
        v16h A1 = load_afrag(ap1 + k);
        v16h B0 = load_afrag(bp0 + k);
        v16h B1 = load_afrag(bp1 + k);
        a00 = wmma_f16_k32(A0, B0, a00);
        a01 = wmma_f16_k32(A0, B1, a01);
        a10 = wmma_f16_k32(A1, B0, a10);
        a11 = wmma_f16_k32(A1, B1, a11);
    }
    // argmax epilogue: lane covers column (lane&15) of each 16-col sub-tile,
    // rows tn + ti*16 + (lane>>4)*8 + r
    int mb = ((lane >> 4) << 3);
#pragma unroll
    for (int tj = 0; tj < 2; ++tj) {
        int col = tm + tj * 16 + (lane & 15);
        float best = -1e30f; int bestn = tn;
#pragma unroll
        for (int ti = 0; ti < 2; ++ti) {
            v8f acc = (ti == 0) ? (tj == 0 ? a00 : a01)
                                : (tj == 0 ? a10 : a11);
            int nb = tn + ti * 16 + mb;
#pragma unroll
            for (int r = 0; r < 8; ++r) {
                int n = nb + r;
                float v = acc[r] * kinv[b * N + n];
                if (v > best) { best = v; bestn = n; }
            }
        }
        unsigned u = __float_as_uint(best);
        unsigned key32 = (u & 0x80000000u) ? ~u : (u | 0x80000000u);
        unsigned long long packed =
            ((unsigned long long)key32 << 32) | (unsigned)bestn;
        atomicMax(keys + (size_t)b * N + col, packed);
    }
}

// ---------------- overlap-add reconstruction of swapped patches -------------
__global__ void k_recon(const unsigned long long* __restrict__ keys,
                        const float* __restrict__ ns, float* __restrict__ R) {
    int i = blockIdx.x * 256 + threadIdx.x;              // B*C*N threads
    if (i >= B * C * N) return;
    int b = i / (C * N);
    int rem = i % (C * N);
    int c = rem / N, pos = rem % N;
    int y = pos / W, x = pos % W;
    const float* nsb = ns + ((size_t)b * C + c) * N;
    const unsigned long long* kb = keys + (size_t)b * N;
    float sum = 0.f;
#pragma unroll
    for (int di = 0; di < 3; ++di) {
        int yy = 1 + y - di; if (yy < 0 || yy >= H) continue;
#pragma unroll
        for (int dj = 0; dj < 3; ++dj) {
            int xx = 1 + x - dj; if (xx < 0 || xx >= W) continue;
            int n = (int)(kb[yy * W + xx] & 0xFFFFFFFFull);
            int iy = n / W + di - 1, ix = n % W + dj - 1;
            if (iy >= 0 && iy < H && ix >= 0 && ix < W) sum += nsb[iy * W + ix];
        }
    }
    int ci = (y == 0 || y == H - 1) ? 2 : 3;
    int cj = (x == 0 || x == W - 1) ? 2 : 3;
    R[i] = sum / (float)(ci * cj);
}

// ======================= host orchestration =================================
extern "C" void kernel_launch(void* const* d_in, const int* in_sizes, int n_in,
                              void* d_out, int out_size, void* d_ws, size_t ws_size,
                              hipStream_t stream) {
    const float* content  = (const float*)d_in[0];
    const float* style    = (const float*)d_in[1];
    const float* strength = (const float*)d_in[2];   // device scalar
    float* out = (float*)d_out;

    char* ws = (char*)d_ws; size_t off = 0;
    auto alloc = [&](size_t bytes) -> void* {
        void* p = ws + off; off = (off + bytes + 255) & ~(size_t)255; return p;
    };
    const size_t FSZ = (size_t)B * C * N * sizeof(float);   // 4.72 MB
    const size_t MSZ = (size_t)B * C * C * sizeof(float);   // 0.5 MB

    float* Fc   = (float*)alloc(FSZ);                // centered content
    float* Fs   = (float*)alloc(FSZ);                // centered style
    float* nc   = (float*)alloc(FSZ);                // whitened content
    float* ns   = (float*)alloc(FSZ);                // whitened style
    float* R    = (float*)alloc(FSZ);                // reassembled
    float* covC = (float*)alloc(MSZ);
    float* covS = (float*)alloc(MSZ);
    float* Yc = (float*)alloc(MSZ); float* Yc2 = (float*)alloc(MSZ);
    float* Zc = (float*)alloc(MSZ); float* Zc2 = (float*)alloc(MSZ);
    float* Tc = (float*)alloc(MSZ);
    float* Ys = (float*)alloc(MSZ); float* Ys2 = (float*)alloc(MSZ);
    float* Zs = (float*)alloc(MSZ); float* Zs2 = (float*)alloc(MSZ);
    float* Ts = (float*)alloc(MSZ);
    float* meanC = (float*)alloc((size_t)B * C * sizeof(float));
    float* meanS = (float*)alloc((size_t)B * C * sizeof(float));
    float* scC   = (float*)alloc(4 * B * sizeof(float));
    float* scS   = (float*)alloc(4 * B * sizeof(float));
    float* ssq   = (float*)alloc((size_t)B * N * sizeof(float));
    float* kinv  = (float*)alloc((size_t)B * N * sizeof(float));
    unsigned long long* keys =
        (unsigned long long*)alloc((size_t)B * N * sizeof(unsigned long long));
    _Float16* PS = (_Float16*)alloc((size_t)B * N * KK * sizeof(_Float16)); // 21 MB
    _Float16* PC = (_Float16*)alloc((size_t)B * N * KK * sizeof(_Float16)); // 21 MB

    dim3 blk32(32), blk256(256);
    dim3 gcov(C / 16, C / 16, B);          // 16x16 tiles of 256x256
    dim3 gapp(N / 16, C / 16, B);          // 144x16 tiles of 256x2304

    // 1) mean-center both features
    k_center<<<dim3(B * C), blk256, 0, stream>>>(content, Fc, meanC);
    k_center<<<dim3(B * C), blk256, 0, stream>>>(style,   Fs, meanS);

    // 2) covariances (f32 WMMA)
    k_cov<<<gcov, blk32, 0, stream>>>(Fc, covC);
    k_cov<<<gcov, blk32, 0, stream>>>(Fs, covS);

    // 3) trace scalars + NS init
    k_trace<<<dim3(B), blk256, 0, stream>>>(covC, scC);
    k_trace<<<dim3(B), blk256, 0, stream>>>(covS, scS);
    int nsElems = B * C * C;
    k_nsinit<<<dim3(nsElems / 256), blk256, 0, stream>>>(covC, scC, Yc, Zc);
    k_nsinit<<<dim3(nsElems / 256), blk256, 0, stream>>>(covS, scS, Ys, Zs);

    // 4) Newton-Schulz: Y -> (A/tr)^{1/2}, Z -> (A/tr)^{-1/2}
    for (int it = 0; it < NS_ITERS; ++it) {
        k_nsT<<<gcov, blk32, 0, stream>>>(Zc, Yc, Tc);
        k_mm256<<<gcov, blk32, 0, stream>>>(Yc, Tc, Yc2);
        k_mm256<<<gcov, blk32, 0, stream>>>(Tc, Zc, Zc2);
        std::swap(Yc, Yc2); std::swap(Zc, Zc2);
        k_nsT<<<gcov, blk32, 0, stream>>>(Zs, Ys, Ts);
        k_mm256<<<gcov, blk32, 0, stream>>>(Ys, Ts, Ys2);
        k_mm256<<<gcov, blk32, 0, stream>>>(Ts, Zs, Zs2);
        std::swap(Ys, Ys2); std::swap(Zs, Zs2);
    }

    // 5) whiten: nc = cov^{-1/2} Fc = (Z/sqrt(tr)) Fc ; same for style
    k_apply<0><<<gapp, blk32, 0, stream>>>(Zc, Fc, scC, 2, nullptr, nullptr,
                                           nullptr, nc);
    k_apply<0><<<gapp, blk32, 0, stream>>>(Zs, Fs, scS, 2, nullptr, nullptr,
                                           nullptr, ns);

    // 6) patch norms on whitened style
    k_ssq<<<dim3((B * N + 255) / 256), blk256, 0, stream>>>(ns, ssq);
    k_knorm<<<dim3((B * N + 255) / 256), blk256, 0, stream>>>(ssq, kinv);

    // 7) f16 patch matrices
    int pth = B * N * C;
    k_patches<<<dim3((pth + 255) / 256), blk256, 0, stream>>>(ns, PS);
    k_patches<<<dim3((pth + 255) / 256), blk256, 0, stream>>>(nc, PC);

    // 8) score GEMM + fused argmax (f16 WMMA, 2x2 blocked, atomic u64 max)
    hipMemsetAsync(keys, 0, (size_t)B * N * sizeof(unsigned long long), stream);
    k_score<<<dim3(N / 32, N / 32, B), blk32, 0, stream>>>(PS, PC, kinv, keys);

    // 9) overlap-add reconstruction + normalization
    int rth = B * C * N;
    k_recon<<<dim3((rth + 255) / 256), blk256, 0, stream>>>(keys, ns, R);

    // 10) center reassembled in place
    k_center<<<dim3(B * C), blk256, 0, stream>>>(R, R, nullptr);

    // 11) color: out = (1-s)*content + s*( sqrt(tr)*Ys @ R + meanS )
    k_apply<1><<<gapp, blk32, 0, stream>>>(Ys, R, scS, 3, meanS, content,
                                           strength, out);
}
